// GRUsolo_59158879535701
// MI455X (gfx1250) — compile-verified, matrix operand order checked by reading, not run on hardware
//
#include <hip/hip_runtime.h>
#include <hip/hip_bf16.h>

// GRU (Keras v2 reset_after) for B=128, T=1024, D=H=512, C=2 on gfx1250.
// Pipeline:
//   k0  init_zero   : zero h0 (f32+bf16) and the global barrier state
//   k1  pack_b x2   : f32 [512,1536] -> bf16 WMMA-B tile layout (W and R)
//   k2  gemm_in     : xp = x@W + b_i  (bf16 WMMA, f32 acc), W staged in LDS
//   k3  gru_scan    : persistent kernel, 1024 steps, R in LDS, grid barrier
//   k4  classifier  : logits = h_last @ Wf + bf

typedef __attribute__((ext_vector_type(16))) __bf16 v16bf;
typedef __attribute__((ext_vector_type(8)))  float  v8f;

#define WMMA_BF16(A, Bm, Cm) \
  __builtin_amdgcn_wmma_f32_16x16x32_bf16(false, (A), false, (Bm), (short)0, (Cm), false, false)

constexpr int kB  = 128;
constexpr int kT  = 1024;
constexpr int kD  = 512;
constexpr int kH  = 512;
constexpr int kH3 = 1536;
constexpr int kNWG = 32;          // workgroups in the persistent scan kernel

// ---------------- workspace layout (bytes) ----------------
constexpr size_t OFF_BAR = 0;                        // 2 u32 (+pad)  256 B
constexpr size_t OFF_HA  = 256;                      // h f32 buf A   256 KB
constexpr size_t OFF_HBA = OFF_HA  + 262144;         // h bf16 buf A  128 KB
constexpr size_t OFF_HB  = OFF_HBA + 131072;         // h f32 buf B   256 KB
constexpr size_t OFF_HBB = OFF_HB  + 262144;         // h bf16 buf B  128 KB
constexpr size_t OFF_WP  = OFF_HBB + 131072;         // W packed bf16 1.5 MB
constexpr size_t OFF_RP  = OFF_WP  + 1572864;        // R packed bf16 1.5 MB
constexpr size_t OFF_XP  = OFF_RP  + 1572864;        // xp bf16 [T,B,3H] 384 MB
constexpr int ZERO_WORDS = (int)((OFF_HB) / 4);      // bar + hA + hbA = 98368 words

// ---------------- k0: deterministic zero-init ----------------
__global__ void init_zero(unsigned* p, int n) {
  int i = blockIdx.x * 256 + threadIdx.x;
  if (i < n) p[i] = 0u;
}

// ---------------- k1: pack f32 [K=512, N=1536] row-major -> bf16 B-tile layout
// tile index = ct*16 + kt (ct<96, kt<16); within tile: lane*16 + i
// element (K = kt*32 + (lane>>4)*16 + i, N = ct*16 + (lane&15))
__global__ __launch_bounds__(256) void pack_b(const float* __restrict__ src,
                                              __bf16* __restrict__ dst) {
  int idx  = blockIdx.x * 256 + threadIdx.x;      // [0, 512*1536)
  int tile = idx >> 9;
  int rem  = idx & 511;
  int lane = rem >> 4;
  int i    = rem & 15;
  int ct   = tile >> 4;
  int kt   = tile & 15;
  int k    = kt * 32 + ((lane >> 4) << 4) + i;
  int n    = ct * 16 + (lane & 15);
  dst[idx] = (__bf16)src[(size_t)k * kH3 + n];
}

// ---------------- k2: xp = x @ W + b_i -> bf16 time-major [T,B,3H]
// grid (1024, 12), block 256 (8 waves). Wave = 1 M-tile x 8 N-tiles, K=512.
// Block's W slice (8 ct x 16 kt tiles = 128 KB) staged through LDS in two
// 64 KB phases; inner loop feeds WMMA from ds_load instead of global.
__global__ __launch_bounds__(256) void gemm_in(const float* __restrict__ x,
                                               const __bf16* __restrict__ Wp,
                                               const float* __restrict__ b_i,
                                               __bf16* __restrict__ xp) {
  __shared__ __bf16 ldsW[8 * 8 * 512];    // 64 KB: [ct 8][kt 8][lane 32][16]

  const int lane = threadIdx.x & 31;
  const int wv   = threadIdx.x >> 5;
  const int mt   = blockIdx.x * 8 + wv;   // [0, 8192) row tiles of M=131072
  const int ctb  = blockIdx.y * 8;        // base col tile, [0, 96)
  const int ln   = lane & 15;
  const int lh   = lane >> 4;

  v8f acc[8];
#pragma unroll
  for (int i = 0; i < 8; ++i) {
    float bi = b_i[(ctb + i) * 16 + ln];
#pragma unroll
    for (int v = 0; v < 8; ++v) acc[i][v] = bi;
  }

  const int    row  = mt * 16 + ln;              // A-row for this lane
  const float* xrow = x + (size_t)row * kD;

  for (int ph = 0; ph < 2; ++ph) {
    __syncthreads();   // previous phase's LDS reads done
    // cooperative stage: 8 ct x 8 kt tiles = 2048 lane-vectors of 16 bf16
    for (int vec = threadIdx.x; vec < 2048; vec += 256) {
      int i   = vec >> 8;        // ct offset 0..7
      int rem = vec & 255;
      int kk  = rem >> 5;        // kt offset 0..7
      int l   = rem & 31;
      size_t tile = (size_t)(ctb + i) * 16 + (ph * 8 + kk);
      *(v16bf*)&ldsW[(size_t)vec * 16] = *(const v16bf*)(Wp + tile * 512 + l * 16);
    }
    __syncthreads();

#pragma unroll 2
    for (int kk = 0; kk < 8; ++kk) {
      const int kt = ph * 8 + kk;
      v16bf a;
      const float4* xv = (const float4*)(xrow + kt * 32 + lh * 16);
#pragma unroll
      for (int q = 0; q < 4; ++q) {
        float4 f = xv[q];
        a[q * 4 + 0] = (__bf16)f.x;
        a[q * 4 + 1] = (__bf16)f.y;
        a[q * 4 + 2] = (__bf16)f.z;
        a[q * 4 + 3] = (__bf16)f.w;
      }
#pragma unroll
      for (int i = 0; i < 8; ++i) {
        v16bf b = *(const v16bf*)&ldsW[((i * 8 + kk) * 32 + lane) * 16];
        acc[i] = WMMA_BF16(a, b, acc[i]);
      }
    }
  }

  // store: C layout -> time-major bf16 xp[t*B*3H + b*3H + col]
#pragma unroll
  for (int i = 0; i < 8; ++i) {
    int col = (ctb + i) * 16 + ln;
#pragma unroll
    for (int v = 0; v < 8; ++v) {
      int r = mt * 16 + v + (lh << 3);   // GEMM row m = b*T + t
      int t = r & (kT - 1);
      int b = r >> 10;
      xp[(size_t)t * (kB * kH3) + (size_t)b * kH3 + col] = (__bf16)acc[i][v];
    }
  }
}

// ---------------- device-wide barrier (32 co-resident WGs) ----------------
__device__ inline void gridBarrier(unsigned* bar) {
  __threadfence();                       // flush this thread's h stores
  __syncthreads();
  if (threadIdx.x == 0) {
    unsigned g = __hip_atomic_load(bar + 1, __ATOMIC_RELAXED, __HIP_MEMORY_SCOPE_AGENT);
    unsigned a = __hip_atomic_fetch_add(bar, 1u, __ATOMIC_ACQ_REL, __HIP_MEMORY_SCOPE_AGENT);
    if (a == (unsigned)(kNWG - 1)) {
      __hip_atomic_store(bar, 0u, __ATOMIC_RELAXED, __HIP_MEMORY_SCOPE_AGENT);
      __hip_atomic_store(bar + 1, g + 1, __ATOMIC_RELEASE, __HIP_MEMORY_SCOPE_AGENT);
    } else {
      while (__hip_atomic_load(bar + 1, __ATOMIC_ACQUIRE, __HIP_MEMORY_SCOPE_AGENT) == g)
        __builtin_amdgcn_s_sleep(1);
    }
  }
  __syncthreads();
  __builtin_amdgcn_fence(__ATOMIC_ACQUIRE, "agent");  // make peers' h visible to all lanes
}

// ---------------- k3: persistent GRU scan ----------------
// grid 32, block 256 (8 waves). WG j owns H-cols [j*16, j*16+16).
// Wave rt owns batch rows [rt*16, rt*16+16). R gate slices live in LDS.
__global__ __launch_bounds__(256) void gru_scan(const __bf16* __restrict__ xp,
                                                const __bf16* __restrict__ Rp,
                                                const float* __restrict__ b_r,
                                                float* hA, float* hB,
                                                __bf16* hbA, __bf16* hbB,
                                                unsigned* bar) {
  __shared__ __bf16 ldsR[3 * 16 * 32 * 16];   // 48 KB: [gate][kt][lane][16]

  const int j    = blockIdx.x;        // col tile of H, [0,32)
  const int lane = threadIdx.x & 31;
  const int rt   = threadIdx.x >> 5;  // row tile, [0,8)
  const int ln   = lane & 15;
  const int lh   = lane >> 4;

  // one-time: stage this WG's three R gate slices (full K) into LDS
  for (int vec = threadIdx.x; vec < 1536; vec += 256) {   // 1536 lane-vectors
    int g   = vec >> 9;
    int rem = vec & 511;
    int kt  = rem >> 5;
    int l   = rem & 31;
    size_t tile = (size_t)((g * 32 + j) * 16 + kt);
    *(v16bf*)&ldsR[(size_t)vec * 16] = *(const v16bf*)(Rp + tile * 512 + l * 16);
  }
  const float brz = b_r[j * 16 + ln];
  const float brr = b_r[512 + j * 16 + ln];
  const float brh = b_r[1024 + j * 16 + ln];
  __syncthreads();

  for (int t = 0; t < kT; ++t) {
    const float*  hcur  = (t & 1) ? hB  : hA;
    float*        hnxt  = (t & 1) ? hA  : hB;
    const __bf16* hbcur = (t & 1) ? hbB : hbA;
    __bf16*       hbnxt = (t & 1) ? hbA : hbB;

    v8f az, ar, ah;
#pragma unroll
    for (int v = 0; v < 8; ++v) { az[v] = brz; ar[v] = brr; ah[v] = brh; }

    const __bf16* hrow = hbcur + (size_t)(rt * 16 + ln) * kH;
    const size_t  xbase = (size_t)t * (kB * kH3);

    // prefetch next step's xp slab for this wave while we compute
    if (t + 1 < kT)
      __builtin_prefetch(xp + xbase + (kB * kH3) + (size_t)(rt * 16 + ln) * kH3 + j * 16, 0, 1);

#pragma unroll 4
    for (int kt = 0; kt < 16; ++kt) {
      v16bf a  = *(const v16bf*)(hrow + kt * 32 + lh * 16);
      v16bf bz = *(const v16bf*)&ldsR[((0  + kt) * 32 + lane) * 16];
      v16bf br = *(const v16bf*)&ldsR[((16 + kt) * 32 + lane) * 16];
      v16bf bh = *(const v16bf*)&ldsR[((32 + kt) * 32 + lane) * 16];
      az = WMMA_BF16(a, bz, az);
      ar = WMMA_BF16(a, br, ar);
      ah = WMMA_BF16(a, bh, ah);
    }

    const int colH = j * 16 + ln;
#pragma unroll
    for (int v = 0; v < 8; ++v) {
      int    r  = rt * 16 + v + (lh << 3);          // batch row
      size_t xo = xbase + (size_t)r * kH3 + colH;
      float xz = (float)xp[xo];
      float xr = (float)xp[xo + 512];
      float xh = (float)xp[xo + 1024];
      float zg = 1.0f / (1.0f + __expf(-(xz + az[v])));
      float rg = 1.0f / (1.0f + __expf(-(xr + ar[v])));
      float hh = tanhf(xh + rg * ah[v]);
      float ho = hcur[(size_t)r * kH + colH];
      float hn = zg * ho + (1.0f - zg) * hh;
      hnxt[(size_t)r * kH + colH]  = hn;
      hbnxt[(size_t)r * kH + colH] = (__bf16)hn;
    }

    gridBarrier(bar);
  }
}

// ---------------- k4: logits = h_last @ Wf + bf ----------------
__global__ __launch_bounds__(256) void classifier(const float* __restrict__ h,
                                                  const float* __restrict__ Wf,
                                                  const float* __restrict__ bfv,
                                                  float* __restrict__ out) {
  int tid = threadIdx.x;          // 256 = 128 batches x 2 classes
  int b = tid >> 1, c = tid & 1;
  float acc = bfv[c];
  for (int k = 0; k < kH; ++k) acc += h[(size_t)b * kH + k] * Wf[k * 2 + c];
  out[b * 2 + c] = acc;
}

extern "C" void kernel_launch(void* const* d_in, const int* in_sizes, int n_in,
                              void* d_out, int out_size, void* d_ws, size_t ws_size,
                              hipStream_t stream) {
  const float* x   = (const float*)d_in[0];
  const float* W   = (const float*)d_in[1];
  const float* R   = (const float*)d_in[2];
  const float* b_i = (const float*)d_in[3];
  const float* b_r = (const float*)d_in[4];
  const float* Wf  = (const float*)d_in[5];
  const float* bf  = (const float*)d_in[6];
  float* out = (float*)d_out;

  char* ws = (char*)d_ws;
  unsigned* bar = (unsigned*)(ws + OFF_BAR);
  float*    hA  = (float*)   (ws + OFF_HA);
  __bf16*   hbA = (__bf16*)  (ws + OFF_HBA);
  float*    hB  = (float*)   (ws + OFF_HB);
  __bf16*   hbB = (__bf16*)  (ws + OFF_HBB);
  __bf16*   Wp  = (__bf16*)  (ws + OFF_WP);
  __bf16*   Rp  = (__bf16*)  (ws + OFF_RP);
  __bf16*   xp  = (__bf16*)  (ws + OFF_XP);

  init_zero<<<(ZERO_WORDS + 255) / 256, 256, 0, stream>>>((unsigned*)ws, ZERO_WORDS);
  pack_b<<<3072, 256, 0, stream>>>(W, Wp);   // 512*1536 / 256
  pack_b<<<3072, 256, 0, stream>>>(R, Rp);
  gemm_in<<<dim3(1024, 12), 256, 0, stream>>>(x, Wp, b_i, xp);
  gru_scan<<<kNWG, 256, 0, stream>>>(xp, Rp, b_r, hA, hB, hbA, hbB, bar);
  classifier<<<1, 256, 0, stream>>>(hA, Wf, bf, out);
}
